// Model_31997506355652
// MI455X (gfx1250) — matrix operand order repaired and checked by reading, not compiled, not run on hardware
//
#include <hip/hip_runtime.h>
#include <hip/hip_bf16.h>
#include <math.h>
#include <stdint.h>

// ---------------------------------------------------------------------------
// MI455X (gfx1250, wave32) implementation, round 4.
// - All GEMMs: v_wmma_f32_16x16x32_f16, f16 operands, f32 accumulate.
// - Operands staged with double-buffered global_load_async_to_lds_b128
//   (ASYNCcnt) when interior/aligned; guarded ds_store fallback at edges.
// - f16 mirrors produced by producer kernels (GEMM dual-store epilogue,
//   softmax, GRU, concat); weights transpose-converted to (N x K) f16 once.
// Workspace requirement: ~225 MB.
// ---------------------------------------------------------------------------

typedef __attribute__((ext_vector_type(16))) _Float16 v16h;
typedef __attribute__((ext_vector_type(8)))  _Float16 v8h;
typedef __attribute__((ext_vector_type(8)))  float    v8f;
typedef __attribute__((ext_vector_type(4)))  int      v4i;

#define BBATCH 32
#define TLEN   200
#define LLEN   25
#define FEATD  500
#define WDIM   300
#define DDIM   512
#define HGRU   256
#define NANCH  7

constexpr float NEGV = -1e10f;

#define GF_BIAS   2
#define GF_SIGM   4
#define GF_TANH   8
#define GF_ADD    16
#define GF_MUL    32
#define GF_C16T   64   // C16 stored transposed: C16[n*ldd + m]

static __host__ __device__ inline int al8(int x) { return (x + 7) & ~7; }

// ---------------- async copy to LDS (CDNA5) ----------------
// Probe-confirmed: builtin exists; param0 = AS1 v4i* (global), param1 = AS3.
__device__ __forceinline__ void async_cp16(_Float16* lds, const _Float16* g) {
#if __has_builtin(__builtin_amdgcn_global_load_async_to_lds_b128)
  __builtin_amdgcn_global_load_async_to_lds_b128(
      (__attribute__((address_space(1))) v4i*)(unsigned long long)(uintptr_t)g,
      (__attribute__((address_space(3))) v4i*)(unsigned)(uintptr_t)lds,
      0, 0);
#else
  asm volatile("global_load_async_to_lds_b128 %0, %1, off"
               :: "v"((unsigned)(uintptr_t)lds),
                  "v"((unsigned long long)(uintptr_t)g)
               : "memory");
#endif
}

__device__ __forceinline__ void async_wait0() {
#if __has_builtin(__builtin_amdgcn_s_wait_asynccnt)
  __builtin_amdgcn_s_wait_asynccnt(0);
#else
  asm volatile("s_wait_asynccnt 0" ::: "memory");
#endif
}

// ---------------------------------------------------------------------------
// GEMM: C = act(alpha * A @ B^T + bias + add) * mul ; optional f16 C16 store.
// A: (M x K) f16, lda; B: (N x K) f16, ldb (k-contiguous both sides).
// Tile 64x128, 8 waves, wave = 32x32 (4 WMMAs / K-step), double-buffered LDS.
// ---------------------------------------------------------------------------
struct GemmP {
  const _Float16* A; const _Float16* B;
  const float* bias; const float* add; const float* mul;
  float* C; _Float16* C16;
  long long sA1, sA2, sB1, sB2, sC1, sC2, sD1, sD2;
  int lda, ldb, ldc, ldd;
  int M, N, K, nb2;
  float alpha; int flags;
};

#define GBM 64
#define GBN 128
#define GBK 32
#define LDSL 40   // padded LDS row stride in halves (80B; 16B-aligned segments)

__global__ __launch_bounds__(256) void k_gemm(GemmP p) {
  __shared__ _Float16 As[2][GBM * LDSL];   // 2 x 5 KB
  __shared__ _Float16 Bs[2][GBN * LDSL];   // 2 x 10 KB

  const int z  = blockIdx.z;
  const int b1 = z / p.nb2;
  const int b2 = z % p.nb2;
  const _Float16* Ab = p.A + b1 * p.sA1 + b2 * p.sA2;
  const _Float16* Bb = p.B + b1 * p.sB1 + b2 * p.sB2;

  const int tm = blockIdx.x * GBM;
  const int tn = blockIdx.y * GBN;
  const int tid  = threadIdx.x;
  const int lane = tid & 31;
  const int w    = tid >> 5;
  const int wm   = w & 1;
  const int wn   = w >> 1;
  const int hsel = lane >> 4;
  const int r0   = lane & 15;

  const int arr  = tid >> 2;
  const int aseg = (tid & 3) * 8;
  const int agm  = tm + arr;
  const _Float16* arow = Ab + (long long)agm * p.lda;
  const int brr  = tid >> 1;
  const int bseg = (tid & 1) * 16;
  const int bgn  = tn + brr;
  const _Float16* brow = Bb + (long long)bgn * p.ldb;

  auto stage = [&](int buf, int k0) {
    {
      const int kb = k0 + aseg;
      _Float16* dst = &As[buf][arr * LDSL + aseg];
      if (agm < p.M && kb + 8 <= p.K) {
        async_cp16(dst, arow + kb);
      } else {
        v8h h = {};
        if (agm < p.M) {
#pragma unroll
          for (int j = 0; j < 8; ++j)
            if (kb + j < p.K) h[j] = arow[kb + j];
        }
        *(v8h*)dst = h;
      }
    }
#pragma unroll
    for (int s = 0; s < 2; ++s) {
      const int kb = k0 + bseg + s * 8;
      _Float16* dst = &Bs[buf][brr * LDSL + bseg + s * 8];
      if (bgn < p.N && kb + 8 <= p.K) {
        async_cp16(dst, brow + kb);
      } else {
        v8h h = {};
        if (bgn < p.N) {
#pragma unroll
          for (int j = 0; j < 8; ++j)
            if (kb + j < p.K) h[j] = brow[kb + j];
        }
        *(v8h*)dst = h;
      }
    }
  };

  v8f acc[2][2] = {};
  const int nk = (p.K + GBK - 1) / GBK;

  stage(0, 0);
  async_wait0();
  __syncthreads();

  for (int t = 0; t < nk; ++t) {
    const int cur = t & 1;
    if (t + 1 < nk) stage(cur ^ 1, (t + 1) * GBK);

    // ---- WMMA from LDS buffer `cur` (EXEC all-ones; uniform flow) ----
    {
      const _Float16* a0p = &As[cur][(wm * 32 + r0) * LDSL];
      v8h a0lo = *(const v8h*)(a0p + hsel * 8);
      v8h a0hi = *(const v8h*)(a0p + 16 + hsel * 8);
      v16h a0 = __builtin_shufflevector(a0lo, a0hi,
                  0,1,2,3,4,5,6,7,8,9,10,11,12,13,14,15);
      const _Float16* a1p = &As[cur][(wm * 32 + 16 + r0) * LDSL];
      v8h a1lo = *(const v8h*)(a1p + hsel * 8);
      v8h a1hi = *(const v8h*)(a1p + 16 + hsel * 8);
      v16h a1 = __builtin_shufflevector(a1lo, a1hi,
                  0,1,2,3,4,5,6,7,8,9,10,11,12,13,14,15);
      const _Float16* b0p = &Bs[cur][(wn * 32 + r0) * LDSL];
      v8h b0lo = *(const v8h*)(b0p + hsel * 8);
      v8h b0hi = *(const v8h*)(b0p + 16 + hsel * 8);
      v16h bf0 = __builtin_shufflevector(b0lo, b0hi,
                  0,1,2,3,4,5,6,7,8,9,10,11,12,13,14,15);
      const _Float16* b1p = &Bs[cur][(wn * 32 + 16 + r0) * LDSL];
      v8h b1lo = *(const v8h*)(b1p + hsel * 8);
      v8h b1hi = *(const v8h*)(b1p + 16 + hsel * 8);
      v16h bf1 = __builtin_shufflevector(b1lo, b1hi,
                  0,1,2,3,4,5,6,7,8,9,10,11,12,13,14,15);

      acc[0][0] = __builtin_amdgcn_wmma_f32_16x16x32_f16(false, a0, false, bf0,
                                                         (short)0, acc[0][0], false, false);
      acc[0][1] = __builtin_amdgcn_wmma_f32_16x16x32_f16(false, a0, false, bf1,
                                                         (short)0, acc[0][1], false, false);
      acc[1][0] = __builtin_amdgcn_wmma_f32_16x16x32_f16(false, a1, false, bf0,
                                                         (short)0, acc[1][0], false, false);
      acc[1][1] = __builtin_amdgcn_wmma_f32_16x16x32_f16(false, a1, false, bf1,
                                                         (short)0, acc[1][1], false, false);
    }
    async_wait0();
    __syncthreads();
  }

  // ---- epilogue ----
  float* Cb = p.C ? p.C + b1 * p.sC1 + b2 * p.sC2 : nullptr;
  _Float16* Db = p.C16 ? p.C16 + b1 * p.sD1 + b2 * p.sD2 : nullptr;
  const float* addb = (p.flags & GF_ADD) ? (p.add + b1 * p.sC1 + b2 * p.sC2) : nullptr;
  const float* mulb = (p.flags & GF_MUL) ? (p.mul + b1 * p.sC1 + b2 * p.sC2) : nullptr;
#pragma unroll
  for (int f = 0; f < 2; ++f) {
#pragma unroll
    for (int g = 0; g < 2; ++g) {
      const int n = tn + wn * 32 + g * 16 + r0;
      const int mbase = tm + wm * 32 + f * 16 + hsel * 8;
      if (n < p.N) {
        const float bn = (p.flags & GF_BIAS) ? p.bias[n] : 0.f;
#pragma unroll
        for (int r = 0; r < 8; ++r) {
          const int m = mbase + r;
          if (m < p.M) {
            float v = acc[f][g][r] * p.alpha + bn;
            if (addb) v += addb[(long long)m * p.ldc + n];
            if (p.flags & GF_SIGM) v = 1.f / (1.f + __expf(-v));
            if (p.flags & GF_TANH) v = tanhf(v);
            if (mulb) v *= mulb[(long long)m * p.ldc + n];
            if (Cb) Cb[(long long)m * p.ldc + n] = v;
            if (Db) {
              const long long di = (p.flags & GF_C16T)
                                 ? ((long long)n * p.ldd + m)
                                 : ((long long)m * p.ldd + n);
              Db[di] = (_Float16)v;
            }
          }
        }
      }
    }
  }
}

// ---------------------------------------------------------------------------
// f32 -> f16 row convert with zero padding: dst (R x dld) = src (R x sld)
// ---------------------------------------------------------------------------
__global__ void k_cvt(const float* src, _Float16* dst, long long R, int cols,
                      int sld, int dld) {
  const int cpr = dld >> 3;
  const long long total = R * cpr;
  for (long long idx = (long long)blockIdx.x * blockDim.x + threadIdx.x;
       idx < total; idx += (long long)gridDim.x * blockDim.x) {
    const long long r = idx / cpr;
    const int c8 = (int)(idx % cpr) * 8;
    const float* s = src + r * sld + c8;
    v8h h = {};
#pragma unroll
    for (int j = 0; j < 8; ++j)
      if (c8 + j < cols) h[j] = (_Float16)s[j];
    *(v8h*)&dst[r * dld + c8] = h;
  }
}

// ---------------------------------------------------------------------------
// f32 -> f16 transpose-convert: dst[z][c*dld + r] = src[z][r*sld + c]
// ---------------------------------------------------------------------------
__global__ void k_trcvt(const float* src, _Float16* dst, int R, int C, int sld,
                        int dld, long long sS, long long sD, int nb) {
  const int cpr = dld >> 3;
  const long long per = (long long)C * cpr;
  const long long total = (long long)nb * per;
  for (long long idx = (long long)blockIdx.x * blockDim.x + threadIdx.x;
       idx < total; idx += (long long)gridDim.x * blockDim.x) {
    const int zz = (int)(idx / per);
    const long long rem = idx % per;
    const int cc = (int)(rem / cpr);
    const int r8 = (int)(rem % cpr) * 8;
    const float* s = src + zz * sS;
    v8h h = {};
#pragma unroll
    for (int j = 0; j < 8; ++j)
      if (r8 + j < R) h[j] = (_Float16)s[(long long)(r8 + j) * sld + cc];
    *(v8h*)&dst[zz * sD + (long long)cc * dld + r8] = h;
  }
}

// ---------------------------------------------------------------------------
// Row softmax with masks; writes fp32 in place and zero-padded f16 copy.
// grid = (Tq, B*nh), block = 128.
// ---------------------------------------------------------------------------
__global__ void k_softmax(float* S, _Float16* S16, const float* kvmask,
                          int Tq, int Tk, int ld16, int nh, int band) {
  float* row = S + ((long long)blockIdx.y * Tq + blockIdx.x) * Tk;
  _Float16* row16 = S16 + ((long long)blockIdx.y * Tq + blockIdx.x) * ld16;
  const int q = blockIdx.x;
  const int b = blockIdx.y / nh;
  const int tid = threadIdx.x;
  __shared__ float shm[4];
  __shared__ float shs[4];
  __shared__ float rM, rS;

  float mx = -3.4e38f;
  for (int k = tid; k < Tk; k += blockDim.x) {
    float v = row[k];
    if (band) { int d = q - k; d = d < 0 ? -d : d; if (d > 3) v += NEGV; }
    if (kvmask) v += (1.f - kvmask[(long long)b * Tk + k]) * NEGV;
    row[k] = v;
    mx = fmaxf(mx, v);
  }
  for (int o = 16; o; o >>= 1) mx = fmaxf(mx, __shfl_xor(mx, o, 32));
  if ((tid & 31) == 0) shm[tid >> 5] = mx;
  __syncthreads();
  if (tid == 0) {
    float m2 = shm[0];
    for (int j = 1; j < 4; ++j) m2 = fmaxf(m2, shm[j]);
    rM = m2;
  }
  __syncthreads();
  mx = rM;

  float sum = 0.f;
  for (int k = tid; k < Tk; k += blockDim.x) {
    float e = __expf(row[k] - mx);
    row[k] = e;
    sum += e;
  }
  for (int o = 16; o; o >>= 1) sum += __shfl_xor(sum, o, 32);
  if ((tid & 31) == 0) shs[tid >> 5] = sum;
  __syncthreads();
  if (tid == 0) rS = shs[0] + shs[1] + shs[2] + shs[3];
  __syncthreads();
  const float inv = 1.f / rS;
  for (int k = tid; k < Tk; k += blockDim.x) {
    const float v = row[k] * inv;
    row[k] = v;
    row16[k] = (_Float16)v;
  }
  for (int k = Tk + tid; k < ld16; k += blockDim.x) row16[k] = (_Float16)0.f;
}

// ---------------------------------------------------------------------------
// mask -> lengths
// ---------------------------------------------------------------------------
__global__ void k_len(const float* mask, int* len, int Tn) {
  __shared__ float sh[4];
  const int b = blockIdx.x, tid = threadIdx.x;
  float s = 0.f;
  for (int i = tid; i < Tn; i += blockDim.x) s += mask[(long long)b * Tn + i];
  for (int o = 16; o; o >>= 1) s += __shfl_xor(s, o, 32);
  if ((tid & 31) == 0) sh[tid >> 5] = s;
  __syncthreads();
  if (tid == 0) {
    float t = 0.f;
    for (int j = 0; j < (int)(blockDim.x >> 5); ++j) t += sh[j];
    len[b] = (int)(t + 0.5f);
  }
}

// ---------------------------------------------------------------------------
// GRU recurrence; writes fp32 out and f16 mirror.
// ---------------------------------------------------------------------------
__global__ __launch_bounds__(256) void k_gru(const float* gx, const float* mask,
                                             const int* lens, const float* w_hh,
                                             const float* b_hh, float* out,
                                             _Float16* out16,
                                             int Tn, int rev, int colOff) {
  const int b = blockIdx.x, tid = threadIdx.x;
  __shared__ float hsh[HGRU];
  hsh[tid] = 0.f;
  __syncthreads();
  const int len = lens[b];
  const float bh0 = b_hh[tid];
  const float bh1 = b_hh[HGRU + tid];
  const float bh2 = b_hh[2 * HGRU + tid];
  const float* w0 = w_hh + (long long)tid * HGRU;
  const float* w1 = w_hh + (long long)(HGRU + tid) * HGRU;
  const float* w2 = w_hh + (long long)(2 * HGRU + tid) * HGRU;

  for (int t = 0; t < Tn; ++t) {
    const int tt = rev ? ((t < len) ? (len - 1 - t) : t) : t;
    float g0 = bh0, g1 = bh1, g2 = bh2;
    for (int k = 0; k < HGRU; ++k) {
      const float hk = hsh[k];
      g0 += w0[k] * hk;
      g1 += w1[k] * hk;
      g2 += w2[k] * hk;
    }
    const float* gxr = gx + ((long long)b * Tn + tt) * (3 * HGRU);
    const float r  = 1.f / (1.f + __expf(-(gxr[tid] + g0)));
    const float zz = 1.f / (1.f + __expf(-(gxr[HGRU + tid] + g1)));
    const float nn = tanhf(gxr[2 * HGRU + tid] + r * g2);
    const float hp = hsh[tid];
    float hn = (1.f - zz) * nn + zz * hp;
    const float mt = mask[(long long)b * Tn + t];
    hn = mt * hn + (1.f - mt) * hp;
    __syncthreads();
    hsh[tid] = hn;
    const long long oi = ((long long)b * Tn + tt) * DDIM + colOff + tid;
    out[oi] = hn;
    out16[oi] = (_Float16)hn;
    __syncthreads();
  }
}

// ---------------------------------------------------------------------------
// Fused tanh-attention.
// ---------------------------------------------------------------------------
__global__ __launch_bounds__(256) void k_v2s(const float* aT, const float* bT,
                                             const float* wst, const float* smat,
                                             const float* node_mask, float* x1) {
  const int bt = blockIdx.x;
  const int b  = bt / TLEN;
  const int tid = threadIdx.x;
  __shared__ float arow[DDIM];
  __shared__ float wsts[DDIM];
  __shared__ float sc[32];
  __shared__ float red[8];

  arow[tid]       = aT[(long long)bt * DDIM + tid];
  arow[256 + tid] = aT[(long long)bt * DDIM + 256 + tid];
  wsts[tid]       = wst[tid];
  wsts[256 + tid] = wst[256 + tid];
  __syncthreads();

  for (int l = 0; l < LLEN; ++l) {
    const float* br = bT + ((long long)b * LLEN + l) * DDIM;
    float p = tanhf(arow[tid] + br[tid]) * wsts[tid]
            + tanhf(arow[256 + tid] + br[256 + tid]) * wsts[256 + tid];
    for (int o = 16; o; o >>= 1) p += __shfl_xor(p, o, 32);
    if ((tid & 31) == 0) red[tid >> 5] = p;
    __syncthreads();
    if (tid == 0) {
      float s2 = 0.f;
      for (int j = 0; j < 8; ++j) s2 += red[j];
      s2 += (1.f - node_mask[b * LLEN + l]) * NEGV;
      sc[l] = s2;
    }
    __syncthreads();
  }
  if (tid == 0) {
    float mx = -3.4e38f;
    for (int l = 0; l < LLEN; ++l) mx = fmaxf(mx, sc[l]);
    float sum = 0.f;
    for (int l = 0; l < LLEN; ++l) { sc[l] = __expf(sc[l] - mx); sum += sc[l]; }
    const float inv = 1.f / sum;
    for (int l = 0; l < LLEN; ++l) sc[l] *= inv;
  }
  __syncthreads();
  float o0 = 0.f, o1 = 0.f;
  for (int l = 0; l < LLEN; ++l) {
    const float* sr = smat + ((long long)b * LLEN + l) * DDIM;
    const float pl = sc[l];
    o0 += pl * sr[tid];
    o1 += pl * sr[256 + tid];
  }
  x1[(long long)bt * DDIM + tid] = o0;
  x1[(long long)bt * DDIM + 256 + tid] = o1;
}

// ---------------------------------------------------------------------------
// concat [v, x1] -> f16 (B*T, 1024)
// ---------------------------------------------------------------------------
__global__ void k_concat16(const float* v, const float* x1, _Float16* cat,
                           long long n) {
  long long i = (long long)blockIdx.x * blockDim.x + threadIdx.x;
  if (i >= n) return;
  const long long m = i >> 10;
  const int c = (int)(i & 1023);
  const float val = (c < DDIM) ? v[m * DDIM + c] : x1[m * DDIM + (c - DDIM)];
  cat[i] = (_Float16)val;
}

// ---------------------------------------------------------------------------
// per-batch argmax + proposal refine + smooth-L1 parts
// ---------------------------------------------------------------------------
__global__ __launch_bounds__(256) void k_box(const float* pf, const float* regb,
                                             const float* fr_label, float* outbox,
                                             float* regpart) {
  const int b = blockIdx.x, tid = threadIdx.x;
  __shared__ float bv[256];
  __shared__ int   bi[256];
  float best = -3.4e38f; int bidx = 0x7fffffff;
  const int TA = TLEN * NANCH;
  for (int i = tid; i < TA; i += 256) {
    const float v = pf[(long long)b * TA + i];
    if (v > best || (v == best && i < bidx)) { best = v; bidx = i; }
  }
  bv[tid] = best; bi[tid] = bidx;
  __syncthreads();
  for (int s = 128; s > 0; s >>= 1) {
    if (tid < s) {
      if (bv[tid + s] > bv[tid] ||
          (bv[tid + s] == bv[tid] && bi[tid + s] < bi[tid])) {
        bv[tid] = bv[tid + s]; bi[tid] = bi[tid + s];
      }
    }
    __syncthreads();
  }
  if (tid == 0) {
    const int idx = bi[0];
    const int t = idx / NANCH, a = idx % NANCH;
    const float widths[NANCH] = {16.f, 32.f, 64.f, 96.f, 128.f, 160.f, 192.f};
    const float w = widths[a];
    const float st = (float)t - 0.5f * (w - 1.f);
    const float en = (float)t + 0.5f * (w - 1.f);
    const float rb0 = st + regb[(long long)b * (TA * 2) + idx * 2 + 0];
    const float rb1 = en + regb[(long long)b * (TA * 2) + idx * 2 + 1];
    outbox[b * 2 + 0] = rb0;
    outbox[b * 2 + 1] = rb1;
    const float d0 = rb0 - fr_label[b * 2 + 0];
    const float d1 = rb1 - fr_label[b * 2 + 1];
    const float a0 = fabsf(d0), a1 = fabsf(d1);
    const float s0 = a0 < 1.f ? 0.5f * d0 * d0 : a0 - 0.5f;
    const float s1 = a1 < 1.f ? 0.5f * d1 * d1 : a1 - 0.5f;
    regpart[b] = s0 + s1;
  }
}

// ---------------------------------------------------------------------------
// BCE + 0.001 * smooth-L1 -> scalar loss
// ---------------------------------------------------------------------------
__global__ __launch_bounds__(256) void k_loss(const float* pf, const float* label,
                                              const float* regpart, float* out) {
  __shared__ float sh[8];
  const int tid = threadIdx.x;
  const int n = BBATCH * TLEN * NANCH;
  float acc = 0.f;
  for (int i = tid; i < n; i += 256) {
    float pc = pf[i];
    pc = fminf(fmaxf(pc, 1e-12f), 1.f - 1e-7f);
    const float lb = label[i];
    acc -= lb * __logf(pc) + (1.f - lb) * __logf(1.f - pc);
  }
  for (int o = 16; o; o >>= 1) acc += __shfl_xor(acc, o, 32);
  if ((tid & 31) == 0) sh[tid >> 5] = acc;
  __syncthreads();
  if (tid == 0) {
    float cls = 0.f;
    for (int j = 0; j < 8; ++j) cls += sh[j];
    cls /= (float)n;
    float rs = 0.f;
    for (int b = 0; b < BBATCH; ++b) rs += regpart[b];
    out[0] = cls + 0.001f * (rs / (float)(BBATCH * 2));
  }
}

// ---------------------------------------------------------------------------
// Host orchestration
// ---------------------------------------------------------------------------
static GemmP G(const _Float16* A, int lda, long long sA1, long long sA2,
               const _Float16* B, int ldb, long long sB1, long long sB2,
               int M, int N, int K, int nb2, float alpha, int flags) {
  GemmP p = {};
  p.A = A; p.B = B; p.lda = lda; p.ldb = ldb;
  p.sA1 = sA1; p.sA2 = sA2; p.sB1 = sB1; p.sB2 = sB2;
  p.M = M; p.N = N; p.K = K; p.nb2 = nb2; p.alpha = alpha; p.flags = flags;
  return p;
}

static void gemm(hipStream_t st, const GemmP& p, int nb1) {
  dim3 g((p.M + GBM - 1) / GBM, (p.N + GBN - 1) / GBN, nb1 * p.nb2);
  k_gemm<<<g, 256, 0, st>>>(p);
}

static void cvt(hipStream_t st, const float* s, _Float16* d, long long R,
                int cols, int sld, int dld) {
  long long total = R * (long long)(dld >> 3);
  int nb = (int)((total + 255) / 256); if (nb > 8192) nb = 8192; if (nb < 1) nb = 1;
  k_cvt<<<nb, 256, 0, st>>>(s, d, R, cols, sld, dld);
}

static void trcvt(hipStream_t st, const float* s, _Float16* d, int R, int C,
                  int sld, int dld, long long sS, long long sD, int nb2) {
  long long total = (long long)nb2 * C * (dld >> 3);
  int nb = (int)((total + 255) / 256); if (nb > 8192) nb = 8192; if (nb < 1) nb = 1;
  k_trcvt<<<nb, 256, 0, st>>>(s, d, R, C, sld, dld, sS, sD, nb2);
}

struct Ctx {
  hipStream_t st;
  void* const* din;
  _Float16* pool[4]; int pidx;
  _Float16 *Q16, *K16, *VT16, *O16, *S16;
  float* S;
};

static _Float16* pnext(Ctx& c) { _Float16* p = c.pool[c.pidx & 3]; c.pidx++; return p; }

static void run_mha(Ctx& c, const float* qin, const _Float16* qin16,
                    const _Float16* kvin16, int Tq, int Tk, int nh, int wbase,
                    const float* kvmask, int band,
                    float* out, _Float16* out16) {
  auto P = [&](int i) { return (const float*)c.din[10 + i]; };
  const int Mq = BBATCH * Tq, Mk = BBATCH * Tk, dh = DDIM / nh;
  const int ldt = al8(Tk);
  // Q projection -> Q16
  {
    _Float16* wq = pnext(c);
    trcvt(c.st, P(wbase + 0), wq, DDIM, DDIM, DDIM, DDIM, 0, 0, 1);
    GemmP p = G(qin16, DDIM, 0, 0, wq, DDIM, 0, 0, Mq, DDIM, DDIM, 1, 1.f, GF_BIAS);
    p.bias = P(wbase + 4); p.C16 = c.Q16; p.ldd = DDIM;
    gemm(c.st, p, 1);
  }
  // K projection -> K16
  {
    _Float16* wk = pnext(c);
    trcvt(c.st, P(wbase + 1), wk, DDIM, DDIM, DDIM, DDIM, 0, 0, 1);
    GemmP p = G(kvin16, DDIM, 0, 0, wk, DDIM, 0, 0, Mk, DDIM, DDIM, 1, 1.f, GF_BIAS);
    p.bias = P(wbase + 5); p.C16 = c.K16; p.ldd = DDIM;
    gemm(c.st, p, 1);
  }
  // V projection -> VT16 (per-batch transposed f16 store: VT[b][d][k])
  {
    _Float16* wv = pnext(c);
    trcvt(c.st, P(wbase + 2), wv, DDIM, DDIM, DDIM, DDIM, 0, 0, 1);
    GemmP p = G(kvin16, DDIM, (long long)Tk * DDIM, 0, wv, DDIM, 0, 0,
                Tk, DDIM, DDIM, 1, 1.f, GF_BIAS | GF_C16T);
    p.bias = P(wbase + 6); p.C16 = c.VT16; p.ldd = ldt;
    p.sD1 = (long long)DDIM * ldt;
    gemm(c.st, p, BBATCH);
  }
  // scores (fp32)
  {
    GemmP p = G(c.Q16, DDIM, (long long)Tq * DDIM, dh,
                c.K16, DDIM, (long long)Tk * DDIM, dh,
                Tq, Tk, dh, nh, 1.f / sqrtf((float)dh), 0);
    p.C = c.S; p.ldc = Tk;
    p.sC1 = (long long)nh * Tq * Tk; p.sC2 = (long long)Tq * Tk;
    gemm(c.st, p, BBATCH);
  }
  k_softmax<<<dim3(Tq, BBATCH * nh), 128, 0, c.st>>>(c.S, c.S16, kvmask,
                                                     Tq, Tk, ldt, nh, band);
  // context: P @ V -> O16
  {
    GemmP p = G(c.S16, ldt, (long long)nh * Tq * ldt, (long long)Tq * ldt,
                c.VT16, ldt, (long long)DDIM * ldt, (long long)dh * ldt,
                Tq, dh, Tk, nh, 1.f, 0);
    p.C16 = c.O16; p.ldd = DDIM;
    p.sD1 = (long long)Tq * DDIM; p.sD2 = dh;
    gemm(c.st, p, BBATCH);
  }
  // output projection + residual -> out (fp32) + out16
  {
    _Float16* wo = pnext(c);
    trcvt(c.st, P(wbase + 3), wo, DDIM, DDIM, DDIM, DDIM, 0, 0, 1);
    GemmP p = G(c.O16, DDIM, 0, 0, wo, DDIM, 0, 0, Mq, DDIM, DDIM, 1, 1.f,
                GF_BIAS | GF_ADD);
    p.bias = P(wbase + 7); p.add = qin;
    p.C = out; p.ldc = DDIM;
    p.C16 = out16; p.ldd = DDIM;
    gemm(c.st, p, 1);
  }
}

extern "C" void kernel_launch(void* const* d_in, const int* in_sizes, int n_in,
                              void* d_out, int out_size, void* d_ws, size_t ws_size,
                              hipStream_t stream) {
  (void)in_sizes; (void)n_in; (void)out_size; (void)ws_size;
  const float* frames     = (const float*)d_in[0];
  const float* frame_mask = (const float*)d_in[1];
  const float* words      = (const float*)d_in[2];
  const float* word_mask  = (const float*)d_in[3];
  const float* label      = (const float*)d_in[4];
  const float* label_mask = (const float*)d_in[5];
  const float* fr_label   = (const float*)d_in[6];
  const float* node_mask  = (const float*)d_in[8];
  auto P = [&](int i) { return (const float*)d_in[10 + i]; };

  const long long NV = (long long)BBATCH * TLEN * DDIM;     // 3,276,800
  const long long NS = (long long)BBATCH * 8 * TLEN * TLEN; // 10,240,000
  const long long NSENT = (long long)BBATCH * LLEN * DDIM;
  const long long NGX = (long long)BBATCH * TLEN * 768;
  const long long NSGX = (long long)BBATCH * LLEN * 768;
  const long long NCAT = (long long)BBATCH * TLEN * 1024;

  float* ws = (float*)d_ws;
  long long off = 0;
  auto alloc = [&](long long n) { float* p = ws + off; off += n; return p; };
  float* vA   = alloc(NV);
  float* vB   = alloc(NV);
  float* Sb   = alloc(NS);
  float* sAb  = alloc(NSENT);
  float* sBb  = alloc(NSENT);
  float* aTb  = alloc(NV);
  float* bTb  = alloc(NSENT);
  float* x1b  = alloc(NV);
  float* gxF  = alloc(NGX);
  float* gxB  = alloc(NGX);
  float* sgF  = alloc(NSGX);
  float* sgB  = alloc(NSGX);
  float* regb = alloc((long long)BBATCH * TLEN * NANCH * 2);
  int* lenF = (int*)alloc(32);
  int* lenW = (int*)alloc(32);
  float* regpart = alloc(32);

  _Float16* hb = (_Float16*)(ws + off);
  long long hoff = 0;
  auto halloc = [&](long long n) {
    _Float16* p = hb + hoff; hoff += (n + 7) & ~7LL; return p;
  };
  _Float16* fr16  = halloc((long long)BBATCH * TLEN * al8(FEATD));
  _Float16* wd16  = halloc((long long)BBATCH * LLEN * al8(WDIM));
  _Float16* v16A  = halloc(NV);
  _Float16* v16B  = halloc(NV);
  _Float16* Q16   = halloc(NV);
  _Float16* K16   = halloc(NV);
  _Float16* VT16  = halloc(NV);
  _Float16* O16   = halloc(NV);
  _Float16* S16   = halloc(NS);
  _Float16* s16A  = halloc(NSENT);
  _Float16* s16B  = halloc(NSENT);
  _Float16* cat16 = halloc(NCAT);
  Ctx c;
  c.st = stream; c.din = d_in; c.pidx = 0;
  for (int i = 0; i < 4; ++i) c.pool[i] = halloc(768 * 1024);
  c.Q16 = Q16; c.K16 = K16; c.VT16 = VT16; c.O16 = O16; c.S16 = S16; c.S = Sb;

  float* outp = (float*)d_out;
  float* boxo = outp;
  float* losso = outp + 64;
  float* pf = outp + 65;

  k_len<<<BBATCH, 128, 0, stream>>>(frame_mask, lenF, TLEN);
  k_len<<<BBATCH, 128, 0, stream>>>(word_mask, lenW, LLEN);

  // one-time input converts (padded f16 mirrors)
  cvt(stream, frames, fr16, (long long)BBATCH * TLEN, FEATD, FEATD, al8(FEATD));
  cvt(stream, words, wd16, (long long)BBATCH * LLEN, WDIM, WDIM, al8(WDIM));

  // ---- VideoEncoder: feature projection ----
  {
    _Float16* wt = pnext(c);
    trcvt(stream, P(0), wt, FEATD, DDIM, DDIM, al8(FEATD), 0, 0, 1);
    GemmP p = G(fr16, al8(FEATD), 0, 0, wt, al8(FEATD), 0, 0,
                BBATCH * TLEN, DDIM, FEATD, 1, 1.f, GF_BIAS);
    p.bias = P(1); p.C = vA; p.ldc = DDIM; p.C16 = v16A; p.ldd = DDIM;
    gemm(stream, p, 1);
  }
  float* vcur = vA; float* valt = vB;
  _Float16* v16cur = v16A; _Float16* v16alt = v16B;

  for (int i = 0; i < 2; ++i) {
    run_mha(c, vcur, v16cur, v16cur, TLEN, TLEN, 8, 2 + 8 * i, nullptr, 1,
            valt, v16alt);
    { float* t = vcur; vcur = valt; valt = t; }
    { _Float16* t = v16cur; v16cur = v16alt; v16alt = t; }
  }

  // ---- vgru ----
  {
    _Float16* wf = pnext(c);
    cvt(stream, P(18), wf, 768, DDIM, DDIM, DDIM);
    GemmP p = G(v16cur, DDIM, 0, 0, wf, DDIM, 0, 0, BBATCH * TLEN, 768, DDIM,
                1, 1.f, GF_BIAS);
    p.bias = P(20); p.C = gxF; p.ldc = 768;
    gemm(stream, p, 1);
  }
  {
    _Float16* wb = pnext(c);
    cvt(stream, P(22), wb, 768, DDIM, DDIM, DDIM);
    GemmP p = G(v16cur, DDIM, 0, 0, wb, DDIM, 0, 0, BBATCH * TLEN, 768, DDIM,
                1, 1.f, GF_BIAS);
    p.bias = P(24); p.C = gxB; p.ldc = 768;
    gemm(stream, p, 1);
  }
  k_gru<<<BBATCH, 256, 0, stream>>>(gxF, frame_mask, lenF, P(19), P(21),
                                    valt, v16alt, TLEN, 0, 0);
  k_gru<<<BBATCH, 256, 0, stream>>>(gxB, frame_mask, lenF, P(23), P(25),
                                    valt, v16alt, TLEN, 1, HGRU);
  { float* t = vcur; vcur = valt; valt = t; }
  { _Float16* t = v16cur; v16cur = v16alt; v16alt = t; }

  // ---- SentenceEncoder ----
  {
    _Float16* wf = pnext(c);
    cvt(stream, P(26), wf, 768, WDIM, WDIM, al8(WDIM));
    GemmP p = G(wd16, al8(WDIM), 0, 0, wf, al8(WDIM), 0, 0, BBATCH * LLEN, 768,
                WDIM, 1, 1.f, GF_BIAS);
    p.bias = P(28); p.C = sgF; p.ldc = 768;
    gemm(stream, p, 1);
  }
  {
    _Float16* wb = pnext(c);
    cvt(stream, P(30), wb, 768, WDIM, WDIM, al8(WDIM));
    GemmP p = G(wd16, al8(WDIM), 0, 0, wb, al8(WDIM), 0, 0, BBATCH * LLEN, 768,
                WDIM, 1, 1.f, GF_BIAS);
    p.bias = P(32); p.C = sgB; p.ldc = 768;
    gemm(stream, p, 1);
  }
  k_gru<<<BBATCH, 256, 0, stream>>>(sgF, word_mask, lenW, P(27), P(29),
                                    sAb, s16A, LLEN, 0, 0);
  k_gru<<<BBATCH, 256, 0, stream>>>(sgB, word_mask, lenW, P(31), P(33),
                                    sAb, s16A, LLEN, 1, HGRU);
  float* scur = sAb; float* salt = sBb;
  _Float16* s16cur = s16A; _Float16* s16alt = s16B;

  // ---- co/self attention stack ----
  for (int i = 0; i < 3; ++i) {
    const int base = 42 + 32 * i;
    run_mha(c, vcur, v16cur, s16cur, TLEN, LLEN, 4, base + 0, node_mask, 0,
            valt, v16alt);
    run_mha(c, scur, s16cur, v16cur, LLEN, TLEN, 4, base + 8, frame_mask, 0,
            salt, s16alt);
    { float* t = vcur; vcur = valt; valt = t; }
    { _Float16* t = v16cur; v16cur = v16alt; v16alt = t; }
    { float* t = scur; scur = salt; salt = t; }
    { _Float16* t = s16cur; s16cur = s16alt; s16alt = t; }
    run_mha(c, vcur, v16cur, v16cur, TLEN, TLEN, 4, base + 16, frame_mask, 0,
            valt, v16alt);
    { float* t = vcur; vcur = valt; valt = t; }
    { _Float16* t = v16cur; v16cur = v16alt; v16alt = t; }
    run_mha(c, scur, s16cur, s16cur, LLEN, LLEN, 4, base + 24, node_mask, 0,
            salt, s16alt);
    { float* t = scur; scur = salt; salt = t; }
    { _Float16* t = s16cur; s16cur = s16alt; s16alt = t; }
  }

  // ---- TanhAttention v2s ----
  {
    _Float16* w1 = pnext(c);
    trcvt(stream, P(138), w1, DDIM, DDIM, DDIM, DDIM, 0, 0, 1);
    GemmP p = G(v16cur, DDIM, 0, 0, w1, DDIM, 0, 0, BBATCH * TLEN, DDIM, DDIM,
                1, 1.f, GF_BIAS);
    p.bias = P(139); p.C = aTb; p.ldc = DDIM;
    gemm(stream, p, 1);
  }
  {
    _Float16* w2 = pnext(c);
    trcvt(stream, P(140), w2, DDIM, DDIM, DDIM, DDIM, 0, 0, 1);
    GemmP p = G(s16cur, DDIM, 0, 0, w2, DDIM, 0, 0, BBATCH * LLEN, DDIM, DDIM,
                1, 1.f, 0);
    p.C = bTb; p.ldc = DDIM;
    gemm(stream, p, 1);
  }
  k_v2s<<<BBATCH * TLEN, 256, 0, stream>>>(aTb, bTb, P(141), scur, node_mask, x1b);

  // ---- fuse + final biGRU ----
  {
    const long long n = (long long)BBATCH * TLEN * 1024;
    k_concat16<<<(unsigned)((n + 255) / 256), 256, 0, stream>>>(vcur, x1b, cat16, n);
  }
  {
    _Float16* wf = pnext(c);
    cvt(stream, P(34), wf, 768, 1024, 1024, 1024);
    GemmP p = G(cat16, 1024, 0, 0, wf, 1024, 0, 0, BBATCH * TLEN, 768, 1024,
                1, 1.f, GF_BIAS);
    p.bias = P(36); p.C = gxF; p.ldc = 768;
    gemm(stream, p, 1);
  }
  {
    _Float16* wb = pnext(c);
    cvt(stream, P(38), wb, 768, 1024, 1024, 1024);
    GemmP p = G(cat16, 1024, 0, 0, wb, 1024, 0, 0, BBATCH * TLEN, 768, 1024,
                1, 1.f, GF_BIAS);
    p.bias = P(40); p.C = gxB; p.ldc = 768;
    gemm(stream, p, 1);
  }
  k_gru<<<BBATCH, 256, 0, stream>>>(gxF, frame_mask, lenF, P(35), P(37),
                                    valt, v16alt, TLEN, 0, 0);
  k_gru<<<BBATCH, 256, 0, stream>>>(gxB, frame_mask, lenF, P(39), P(41),
                                    valt, v16alt, TLEN, 1, HGRU);
  _Float16* x16fin = v16alt;

  // ---- heads ----
  {
    _Float16* wf = pnext(c);
    trcvt(stream, P(142), wf, DDIM, NANCH, NANCH, DDIM, 0, 0, 1);
    GemmP p = G(x16fin, DDIM, 0, 0, wf, DDIM, 0, 0, BBATCH * TLEN, NANCH, DDIM,
                1, 1.f, GF_BIAS | GF_SIGM | GF_MUL);
    p.bias = P(143); p.mul = label_mask; p.C = pf; p.ldc = NANCH;
    gemm(stream, p, 1);
  }
  {
    _Float16* wr = pnext(c);
    trcvt(stream, P(144), wr, DDIM, 2 * NANCH, 2 * NANCH, DDIM, 0, 0, 1);
    GemmP p = G(x16fin, DDIM, 0, 0, wr, DDIM, 0, 0, BBATCH * TLEN, 2 * NANCH,
                DDIM, 1, 1.f, GF_BIAS);
    p.bias = P(145); p.C = regb; p.ldc = 2 * NANCH;
    gemm(stream, p, 1);
  }

  // ---- argmax box + losses ----
  k_box<<<BBATCH, 256, 0, stream>>>(pf, regb, fr_label, boxo, regpart);
  k_loss<<<1, 256, 0, stream>>>(pf, label, regpart, losso);
}
